// MoELayer_84593675862651
// MI455X (gfx1250) — compile-verified, matrix operand order checked by reading, not run on hardware
//
#include <hip/hip_runtime.h>
#include <hip/hip_bf16.h>
#include <math.h>

typedef __bf16 bf16_t;
typedef __attribute__((ext_vector_type(16))) __bf16 v16bf;
typedef __attribute__((ext_vector_type(8)))  __bf16 v8bf;
typedef __attribute__((ext_vector_type(4)))  __bf16 v4bf;
typedef __attribute__((ext_vector_type(8)))  float  v8f;

constexpr int DIM  = 1024;
constexpr int FF   = 4096;
constexpr int NTOK = 8192;   // 4 * 2048
constexpr int NEXP = 8;

// ---------------------------------------------------------------------------
// 16-bit operand loader per ISA 7.12.2 (A 16x32 / B 32x16 are symmetric):
// lane: half = lane>>4, row = lane&15. 16 bf16 elements per lane:
//   elems 0..7  -> K = k0 + half*8      + (0..7)
//   elems 8..15 -> K = k0 + 16 + half*8 + (0..7)
// Two contiguous 16B loads.
// ---------------------------------------------------------------------------
__device__ __forceinline__ v16bf load_op_bf16(const bf16_t* __restrict__ row,
                                              int k0, int half) {
  v8bf lo = *reinterpret_cast<const v8bf*>(row + k0 + half * 8);
  v8bf hi = *reinterpret_cast<const v8bf*>(row + k0 + 16 + half * 8);
  v16bf a;
#pragma unroll
  for (int i = 0; i < 8; ++i) { a[i] = lo[i]; a[i + 8] = hi[i]; }
  return a;
}

// Fallback loader from an fp32 weight row; converts to bf16 in-register.
__device__ __forceinline__ v16bf load_op_f32(const float* __restrict__ row,
                                             int k0, int half) {
  const float* p0 = row + k0 + half * 8;
  const float* p1 = row + k0 + 16 + half * 8;
  float4 c0 = *reinterpret_cast<const float4*>(p0);
  float4 c1 = *reinterpret_cast<const float4*>(p0 + 4);
  float4 c2 = *reinterpret_cast<const float4*>(p1);
  float4 c3 = *reinterpret_cast<const float4*>(p1 + 4);
  v16bf b;
  b[0]  = (bf16_t)c0.x; b[1]  = (bf16_t)c0.y; b[2]  = (bf16_t)c0.z; b[3]  = (bf16_t)c0.w;
  b[4]  = (bf16_t)c1.x; b[5]  = (bf16_t)c1.y; b[6]  = (bf16_t)c1.z; b[7]  = (bf16_t)c1.w;
  b[8]  = (bf16_t)c2.x; b[9]  = (bf16_t)c2.y; b[10] = (bf16_t)c2.z; b[11] = (bf16_t)c2.w;
  b[12] = (bf16_t)c3.x; b[13] = (bf16_t)c3.y; b[14] = (bf16_t)c3.z; b[15] = (bf16_t)c3.w;
  return b;
}

template <bool WBF16>
__device__ __forceinline__ v16bf load_w(const void* __restrict__ base,
                                        size_t row, size_t stride,
                                        int k0, int half) {
  if constexpr (WBF16)
    return load_op_bf16((const bf16_t*)base + row * stride, k0, half);
  else
    return load_op_f32((const float*)base + row * stride, k0, half);
}

// ---------------------------------------------------------------------------
// Router: one wave per token; top-2-of-softmax (full denominator cancels).
// ---------------------------------------------------------------------------
__global__ void __launch_bounds__(256) router_kernel(
    const float* __restrict__ x, const float* __restrict__ Wr,
    float* __restrict__ gate) {
  const int wave = threadIdx.x >> 5;
  const int lane = threadIdx.x & 31;
  const int t = blockIdx.x * 8 + wave;
  const float* xr = x + (size_t)t * DIM;

  float logit[NEXP];
#pragma unroll
  for (int e = 0; e < NEXP; ++e) {
    const float* wr = Wr + e * DIM;
    float s = 0.f;
    for (int d = lane; d < DIM; d += 32) s += xr[d] * wr[d];
#pragma unroll
    for (int off = 16; off >= 1; off >>= 1) s += __shfl_xor(s, off, 32);
    logit[e] = s;  // replicated across lanes
  }

  float v1 = -1e30f; int i1 = 0;
#pragma unroll
  for (int e = 0; e < NEXP; ++e)
    if (logit[e] > v1) { v1 = logit[e]; i1 = e; }
  float v2 = -1e30f; int i2 = 0;
#pragma unroll
  for (int e = 0; e < NEXP; ++e)
    if (e != i1 && logit[e] > v2) { v2 = logit[e]; i2 = e; }

  float p2 = expf(v2 - v1);
  float w1 = 1.f / (1.f + p2);
  float w2 = p2 * w1;
  if (lane < NEXP)
    gate[(size_t)t * NEXP + lane] = (lane == i1) ? w1 : ((lane == i2) ? w2 : 0.f);
}

// fp32 -> bf16, 4 elems/thread (used for x and, when ws allows, W1/W2)
__global__ void __launch_bounds__(256) cvt_bf16_kernel(
    const float* __restrict__ src, bf16_t* __restrict__ dst) {
  size_t i = ((size_t)blockIdx.x * blockDim.x + threadIdx.x) * 4;
  float4 v = *reinterpret_cast<const float4*>(src + i);
  v4bf o;
  o[0] = (bf16_t)v.x; o[1] = (bf16_t)v.y; o[2] = (bf16_t)v.z; o[3] = (bf16_t)v.w;
  *reinterpret_cast<v4bf*>(dst + i) = o;
}

__global__ void __launch_bounds__(256) zero_out_kernel(float* __restrict__ out) {
  size_t i = ((size_t)blockIdx.x * blockDim.x + threadIdx.x) * 4;
  float4 z = {0.f, 0.f, 0.f, 0.f};
  *reinterpret_cast<float4*>(out + i) = z;
}

__global__ void zero_cnt_kernel(int* __restrict__ cnt) {
  if (threadIdx.x < NEXP) cnt[threadIdx.x] = 0;
}

// Token -> per-expert compacted index lists (routed = 4x fewer FLOPs vs dense).
__global__ void __launch_bounds__(256) compact_kernel(
    const float* __restrict__ gate, int* __restrict__ cnt,
    int* __restrict__ idx) {
  int t = blockIdx.x * blockDim.x + threadIdx.x;
  const float* g = gate + (size_t)t * NEXP;
#pragma unroll
  for (int e = 0; e < NEXP; ++e) {
    if (g[e] > 0.f) {
      int pos = atomicAdd(&cnt[e], 1);
      idx[(size_t)e * NTOK + pos] = t;
    }
  }
}

// ---------------------------------------------------------------------------
// GEMM1: h[pos, f] = GELU( xb[idx[pos]] . W1e[f] + b1e[f] ), bf16 out.
// Wave tile 32M x 64N (8 WMMA / K-step, B reused 2x, A 4x).
// Block = 8 waves = 32M x 512N; grid (FF/512, NTOK/32).
// Row indices clamped so EXEC stays all-ones through every WMMA.
// ---------------------------------------------------------------------------
template <bool WBF16>
__global__ void __launch_bounds__(256) gemm1_kernel(
    const bf16_t* __restrict__ xb, const void* __restrict__ W1e,
    const float* __restrict__ b1e, const int* __restrict__ idx_e,
    const int* __restrict__ cnt_e, bf16_t* __restrict__ h) {
  const int cnt = *cnt_e;
  const int m0 = blockIdx.y * 32;
  if (m0 >= cnt) return;

  const int lane = threadIdx.x & 31;
  const int wave = threadIdx.x >> 5;
  const int half = lane >> 4;
  const int r    = lane & 15;
  const int n0   = blockIdx.x * 512 + wave * 64;

  const bf16_t* arow0 = xb + (size_t)idx_e[min(m0 + r,      cnt - 1)] * DIM;
  const bf16_t* arow1 = xb + (size_t)idx_e[min(m0 + 16 + r, cnt - 1)] * DIM;

  v8f acc[2][4];
#pragma unroll
  for (int m = 0; m < 2; ++m)
#pragma unroll
    for (int nn = 0; nn < 4; ++nn)
#pragma unroll
      for (int i = 0; i < 8; ++i) acc[m][nn][i] = 0.f;

  for (int kk = 0; kk < DIM / 32; ++kk) {
    const int k0 = kk * 32;
    v16bf a0 = load_op_bf16(arow0, k0, half);
    v16bf a1 = load_op_bf16(arow1, k0, half);
#pragma unroll
    for (int nn = 0; nn < 4; ++nn) {
      v16bf b = load_w<WBF16>(W1e, (size_t)(n0 + nn * 16 + r), DIM, k0, half);
      acc[0][nn] = __builtin_amdgcn_wmma_f32_16x16x32_bf16(
          false, a0, false, b, (short)0, acc[0][nn], false, false);
      acc[1][nn] = __builtin_amdgcn_wmma_f32_16x16x32_bf16(
          false, a1, false, b, (short)0, acc[1][nn], false, false);
    }
  }

#pragma unroll
  for (int m = 0; m < 2; ++m)
#pragma unroll
    for (int nn = 0; nn < 4; ++nn) {
      const int f = n0 + nn * 16 + r;
      const float bias = b1e[f];
#pragma unroll
      for (int i = 0; i < 8; ++i) {
        const int pos = m0 + 16 * m + i + 8 * half;  // < NTOK; rows>=cnt unused
        float v = acc[m][nn][i] + bias;
        float g = 0.5f * v * (1.0f + erff(v * 0.70710678118654752f));
        h[(size_t)pos * FF + f] = (bf16_t)g;
      }
    }
}

// ---------------------------------------------------------------------------
// GEMM2: out[idx[pos], d] += gate[idx[pos], e] * (h[pos] . W2e[d] + b2e[d]).
// Each token appears at most once per expert list; experts run sequentially
// on the stream -> plain read-modify-write is race-free.
// ---------------------------------------------------------------------------
template <bool WBF16>
__global__ void __launch_bounds__(256) gemm2_kernel(
    const bf16_t* __restrict__ h, const void* __restrict__ W2e,
    const float* __restrict__ b2e, const float* __restrict__ gate,
    const int* __restrict__ idx_e, const int* __restrict__ cnt_e,
    float* __restrict__ out, int e) {
  const int cnt = *cnt_e;
  const int m0 = blockIdx.y * 32;
  if (m0 >= cnt) return;

  const int lane = threadIdx.x & 31;
  const int wave = threadIdx.x >> 5;
  const int half = lane >> 4;
  const int r    = lane & 15;
  const int n0   = blockIdx.x * 512 + wave * 64;

  const bf16_t* arow0 = h + (size_t)min(m0 + r,      cnt - 1) * FF;
  const bf16_t* arow1 = h + (size_t)min(m0 + 16 + r, cnt - 1) * FF;

  v8f acc[2][4];
#pragma unroll
  for (int m = 0; m < 2; ++m)
#pragma unroll
    for (int nn = 0; nn < 4; ++nn)
#pragma unroll
      for (int i = 0; i < 8; ++i) acc[m][nn][i] = 0.f;

  for (int kk = 0; kk < FF / 32; ++kk) {
    const int k0 = kk * 32;
    v16bf a0 = load_op_bf16(arow0, k0, half);
    v16bf a1 = load_op_bf16(arow1, k0, half);
#pragma unroll
    for (int nn = 0; nn < 4; ++nn) {
      v16bf b = load_w<WBF16>(W2e, (size_t)(n0 + nn * 16 + r), FF, k0, half);
      acc[0][nn] = __builtin_amdgcn_wmma_f32_16x16x32_bf16(
          false, a0, false, b, (short)0, acc[0][nn], false, false);
      acc[1][nn] = __builtin_amdgcn_wmma_f32_16x16x32_bf16(
          false, a1, false, b, (short)0, acc[1][nn], false, false);
    }
  }

#pragma unroll
  for (int m = 0; m < 2; ++m)
#pragma unroll
    for (int nn = 0; nn < 4; ++nn) {
      const int d = n0 + nn * 16 + r;
      const float bias = b2e[d];
#pragma unroll
      for (int i = 0; i < 8; ++i) {
        const int pos = m0 + 16 * m + i + 8 * half;
        if (pos < cnt) {
          const int t = idx_e[pos];
          const float g = gate[(size_t)t * NEXP + e];
          out[(size_t)t * DIM + d] += g * (acc[m][nn][i] + bias);
        }
      }
    }
}

// ---------------------------------------------------------------------------
extern "C" void kernel_launch(void* const* d_in, const int* in_sizes, int n_in,
                              void* d_out, int out_size, void* d_ws, size_t ws_size,
                              hipStream_t stream) {
  const float* x  = (const float*)d_in[0];
  const float* Wr = (const float*)d_in[1];
  const float* W1 = (const float*)d_in[2];
  const float* b1 = (const float*)d_in[3];
  const float* W2 = (const float*)d_in[4];
  const float* b2 = (const float*)d_in[5];
  float* out = (float*)d_out;

  char* ws = (char*)d_ws;
  size_t off = 0;
  float*  gate = (float*)(ws + off); off += (size_t)NTOK * NEXP * sizeof(float);
  int*    cnt  = (int*)(ws + off);   off += 4096;
  int*    idx  = (int*)(ws + off);   off += (size_t)NTOK * NEXP * sizeof(int);
  bf16_t* xb   = (bf16_t*)(ws + off); off += (size_t)NTOK * DIM * sizeof(bf16_t);
  bf16_t* h    = (bf16_t*)(ws + off); off += (size_t)NTOK * FF * sizeof(bf16_t);
  // Optional bf16 weight arenas (used only if workspace is large enough).
  bf16_t* w1b  = (bf16_t*)(ws + off); off += (size_t)NEXP * FF * DIM * sizeof(bf16_t);
  bf16_t* w2b  = (bf16_t*)(ws + off); off += (size_t)NEXP * DIM * FF * sizeof(bf16_t);
  const bool preconv = (ws_size >= off);  // ws_size constant -> deterministic

  router_kernel<<<NTOK / 8, 256, 0, stream>>>(x, Wr, gate);
  zero_out_kernel<<<(NTOK * DIM) / (256 * 4), 256, 0, stream>>>(out);
  cvt_bf16_kernel<<<(NTOK * DIM) / (256 * 4), 256, 0, stream>>>(x, xb);
  zero_cnt_kernel<<<1, 32, 0, stream>>>(cnt);
  compact_kernel<<<NTOK / 256, 256, 0, stream>>>(gate, cnt, idx);

  if (preconv) {
    const size_t wElems = (size_t)NEXP * FF * DIM;  // 33.5M each
    cvt_bf16_kernel<<<(int)(wElems / (256 * 4)), 256, 0, stream>>>(W1, w1b);
    cvt_bf16_kernel<<<(int)(wElems / (256 * 4)), 256, 0, stream>>>(W2, w2b);
  }

  for (int e = 0; e < NEXP; ++e) {
    const float* b1e = b1 + (size_t)e * FF;
    const float* b2e = b2 + (size_t)e * DIM;
    const int* idx_e = idx + (size_t)e * NTOK;
    const dim3 g1(FF / 512, NTOK / 32), g2(DIM / 512, NTOK / 32);
    if (preconv) {
      gemm1_kernel<true><<<g1, 256, 0, stream>>>(
          xb, w1b + (size_t)e * FF * DIM, b1e, idx_e, cnt + e, h);
      gemm2_kernel<true><<<g2, 256, 0, stream>>>(
          h, w2b + (size_t)e * DIM * FF, b2e, gate, idx_e, cnt + e, out, e);
    } else {
      gemm1_kernel<false><<<g1, 256, 0, stream>>>(
          xb, W1 + (size_t)e * FF * DIM, b1e, idx_e, cnt + e, h);
      gemm2_kernel<false><<<g2, 256, 0, stream>>>(
          h, W2 + (size_t)e * DIM * FF, b2e, gate, idx_e, cnt + e, out, e);
    }
  }
}
